// Sage_Classifier_49323404427444
// MI455X (gfx1250) — matrix operand order repaired and checked by reading, not compiled, not run on hardware
//
#include <hip/hip_runtime.h>
#include <hip/hip_bf16.h>

#define N_NODES  50000
#define N_EDGES  800000
#define NEMBED   128
#define HDIM     256
#define NCLASS   40
#define NCLASS_PAD 48

#define MTILE 32     // node rows per block in the fused GEMM kernel
#define APAD  136    // LDS row stride (halves) for 128-wide A tiles (conflict-free b128)
#define HPAD  264    // LDS row stride (halves) for 256-wide h tile

typedef __attribute__((ext_vector_type(16))) _Float16 v16h;
typedef __attribute__((ext_vector_type(8)))  _Float16 v8h;
typedef __attribute__((ext_vector_type(4)))  _Float16 v4h;
typedef __attribute__((ext_vector_type(8)))  float    v8f;

// ---------------- workspace layout (bytes) ----------------
// agg   : [0, 25,600,000)                50000*128 f32
// deg   : [25,600,000, 25,800,000)       50000 f32
// Wl16  : [25,800,192, +65,536)          256*128 f16
// Wr16  : [25,865,728, +65,536)          256*128 f16
// Wm16  : [25,931,264, +24,576)          48*256 f16 (rows 40..47 zero)
static const size_t OFF_AGG = 0;
static const size_t OFF_DEG = (size_t)N_NODES * NEMBED * 4;          // 25,600,000
static const size_t OFF_WL  = 25800192;
static const size_t OFF_WR  = OFF_WL + (size_t)HDIM * NEMBED * 2;
static const size_t OFF_WM  = OFF_WR + (size_t)HDIM * NEMBED * 2;

// Native CDNA5 no-return f32 atomic add (tracked by STOREcnt; s_endpgm waits idle).
// Inline asm guarantees global_atomic_add_f32 instead of a possible CAS-loop
// expansion of the generic atomic builtin.
__device__ __forceinline__ void atomic_add_f32_native(float* p, float v) {
  asm volatile("global_atomic_add_f32 %0, %1, off" : : "v"(p), "v"(v) : "memory");
}

// ---------------- kernels ----------------

__global__ void k_zero(float* __restrict__ p, int n) {
  int i = blockIdx.x * blockDim.x + threadIdx.x;
  int stride = gridDim.x * blockDim.x;
  for (; i < n; i += stride) p[i] = 0.0f;
}

__global__ void k_cvt_w(const float* __restrict__ Wl, const float* __restrict__ Wr,
                        const float* __restrict__ Wm,
                        _Float16* __restrict__ Wl16, _Float16* __restrict__ Wr16,
                        _Float16* __restrict__ Wm16) {
  int i = blockIdx.x * blockDim.x + threadIdx.x;
  if (i < HDIM * NEMBED) {
    Wl16[i] = (_Float16)Wl[i];
    Wr16[i] = (_Float16)Wr[i];
  }
  if (i < NCLASS_PAD * HDIM) {
    int n = i >> 8;  // / HDIM
    Wm16[i] = (n < NCLASS) ? (_Float16)Wm[i] : (_Float16)0.0f;
  }
}

// One wave (32 lanes) per edge: lane l handles floats [4l, 4l+4) of the 128-wide row.
// Gather is one coalesced 512B row read (L2-resident x); scatter is 4 native
// f32 atomics per lane into the L2-resident agg buffer.
__global__ void k_scatter(const float* __restrict__ x, const int* __restrict__ ei,
                          float* __restrict__ agg, float* __restrict__ deg) {
  int gid = blockIdx.x * blockDim.x + threadIdx.x;
  int e = gid >> 5;
  int lane = threadIdx.x & 31;
  if (e >= N_EDGES) return;
  int src = ei[e];
  int dst = ei[N_EDGES + e];
  float4 v = ((const float4*)(x + (size_t)src * NEMBED))[lane];
  float* a = agg + (size_t)dst * NEMBED + lane * 4;
  atomic_add_f32_native(a + 0, v.x);
  atomic_add_f32_native(a + 1, v.y);
  atomic_add_f32_native(a + 2, v.z);
  atomic_add_f32_native(a + 3, v.w);
  if (lane == 0)
    atomic_add_f32_native(deg + dst, 1.0f);
}

// A fragment (16x32 f16): halves 0..7 = K(k0+aoff .. +7), halves 8..15 = K(k0+16+aoff .. +7)
__device__ __forceinline__ v16h ldA(const _Float16* p /*row base + k0*/, int aoff) {
  union { v16h v; v8h h[2]; } u;
  u.h[0] = *(const v8h*)(p + aoff);
  u.h[1] = *(const v8h*)(p + 16 + aoff);
  return u.v;
}
// B fragment (32x16 f16): halves 0..7 = K(k0+boff .. +7), halves 8..15 = K(k0+16+boff .. +7) for column n=lane%16
__device__ __forceinline__ v16h ldB(const _Float16* p /*row base + k0 + boff*/) {
  union { v16h v; v8h h[2]; } u;
  u.h[0] = *(const v8h*)(p);
  u.h[1] = *(const v8h*)(p + 16);
  return u.v;
}

__launch_bounds__(256)
__global__ void k_sage_fused(const float* __restrict__ x,
                             const float* __restrict__ agg,
                             const float* __restrict__ deg,
                             const _Float16* __restrict__ Wl16,
                             const _Float16* __restrict__ Wr16,
                             const _Float16* __restrict__ Wm16,
                             const float* __restrict__ b_l,
                             const float* __restrict__ b_mlp,
                             float* __restrict__ out) {
  __shared__ _Float16 ldsM[MTILE * APAD];  // mean tile, f16
  __shared__ _Float16 ldsX[MTILE * APAD];  // x tile, f16
  __shared__ _Float16 ldsH[MTILE * HPAD];  // relu hidden tile, f16

  const int tid  = threadIdx.x;
  const int row0 = blockIdx.x * MTILE;

  // ---- stage A tiles: mean = agg/max(deg,1), x -> f16 LDS ----
  {
    int cv = tid & 31;   // float4 column group: cols [4cv, 4cv+4)
    int r8 = tid >> 5;   // 0..7
    for (int it = 0; it < MTILE / 8; ++it) {
      int lr = it * 8 + r8;
      int gr = row0 + lr;
      if (gr >= N_NODES) gr = N_NODES - 1;  // clamp; stores are masked later
      float4 xv = ((const float4*)(x   + (size_t)gr * NEMBED))[cv];
      float4 av = ((const float4*)(agg + (size_t)gr * NEMBED))[cv];
      float rd = 1.0f / fmaxf(deg[gr], 1.0f);
      v4h hx = { (_Float16)xv.x, (_Float16)xv.y, (_Float16)xv.z, (_Float16)xv.w };
      v4h hm = { (_Float16)(av.x * rd), (_Float16)(av.y * rd),
                 (_Float16)(av.z * rd), (_Float16)(av.w * rd) };
      *(v4h*)&ldsX[lr * APAD + cv * 4] = hx;
      *(v4h*)&ldsM[lr * APAD + cv * 4] = hm;
    }
  }
  __syncthreads();

  const int wave = tid >> 5;
  const int lane = tid & 31;
  const int nl   = lane & 15;     // A row (M) / B column (N) within tile
  const int hi   = lane >> 4;     // half-wave select
  const int aoff = hi * 8;        // A K sub-offset
  const int boff = hi * 8;        // B K sub-offset

  // ---- GEMM 1: h = relu(mean @ Wl^T + b_l + x @ Wr^T), 16 N-tiles over 8 waves x 2 ----
  for (int pass = 0; pass < 2; ++pass) {
    const int nt   = wave + pass * 8;     // HDIM tile 0..15
    const int ncol = nt * 16 + nl;        // output column 0..255
    const _Float16* wl = Wl16 + (size_t)ncol * NEMBED;
    const _Float16* wr = Wr16 + (size_t)ncol * NEMBED;
    v8f acc0 = {};   // rows 0..15 of the block tile
    v8f acc1 = {};   // rows 16..31
    for (int k0 = 0; k0 < NEMBED; k0 += 32) {
      v16h bl = ldB(wl + k0 + boff);
      v16h br = ldB(wr + k0 + boff);
      v16h am0 = ldA(&ldsM[(0  + nl) * APAD + k0], aoff);
      v16h ax0 = ldA(&ldsX[(0  + nl) * APAD + k0], aoff);
      v16h am1 = ldA(&ldsM[(16 + nl) * APAD + k0], aoff);
      v16h ax1 = ldA(&ldsX[(16 + nl) * APAD + k0], aoff);
      acc0 = __builtin_amdgcn_wmma_f32_16x16x32_f16(false, am0, false, bl, (short)0, acc0, false, false);
      acc0 = __builtin_amdgcn_wmma_f32_16x16x32_f16(false, ax0, false, br, (short)0, acc0, false, false);
      acc1 = __builtin_amdgcn_wmma_f32_16x16x32_f16(false, am1, false, bl, (short)0, acc1, false, false);
      acc1 = __builtin_amdgcn_wmma_f32_16x16x32_f16(false, ax1, false, br, (short)0, acc1, false, false);
    }
    float blv = b_l[ncol];
    #pragma unroll
    for (int r = 0; r < 8; ++r) {
      int m0 = r + hi * 8;  // C/D layout: row = r + 8*(lane>=16)
      float h0 = fmaxf(acc0[r] + blv, 0.0f);
      float h1 = fmaxf(acc1[r] + blv, 0.0f);
      ldsH[m0 * HPAD + ncol]        = (_Float16)h0;
      ldsH[(16 + m0) * HPAD + ncol] = (_Float16)h1;
    }
  }
  __syncthreads();

  // ---- GEMM 2: out = h @ Wm^T + b_mlp, 2 M-subtiles x 3 N-tiles (48 padded cols) ----
  for (int t = wave; t < 6; t += 8) {
    int msub = t / 3;
    int n2   = t - msub * 3;
    int ncol = n2 * 16 + nl;   // 0..47
    const _Float16* wm = Wm16 + (size_t)ncol * HDIM;
    v8f acc = {};
    for (int k0 = 0; k0 < HDIM; k0 += 32) {
      v16h b = ldB(wm + k0 + boff);
      v16h a = ldA(&ldsH[(msub * 16 + nl) * HPAD + k0], aoff);
      acc = __builtin_amdgcn_wmma_f32_16x16x32_f16(false, a, false, b, (short)0, acc, false, false);
    }
    if (ncol < NCLASS) {
      float bm = b_mlp[ncol];
      #pragma unroll
      for (int r = 0; r < 8; ++r) {
        int row = row0 + msub * 16 + r + hi * 8;
        if (row < N_NODES)
          out[(size_t)row * NCLASS + ncol] = acc[r] + bm;
      }
    }
  }
}

// ---------------- host launch ----------------
extern "C" void kernel_launch(void* const* d_in, const int* in_sizes, int n_in,
                              void* d_out, int out_size, void* d_ws, size_t ws_size,
                              hipStream_t stream) {
  const float* x     = (const float*)d_in[0];
  const float* W_l   = (const float*)d_in[1];
  const float* b_l   = (const float*)d_in[2];
  const float* W_r   = (const float*)d_in[3];
  const float* W_mlp = (const float*)d_in[4];
  const float* b_mlp = (const float*)d_in[5];
  const int*   ei    = (const int*)d_in[6];
  float* out = (float*)d_out;

  char* ws = (char*)d_ws;
  float*     agg  = (float*)(ws + OFF_AGG);
  float*     deg  = (float*)(ws + OFF_DEG);
  _Float16*  Wl16 = (_Float16*)(ws + OFF_WL);
  _Float16*  Wr16 = (_Float16*)(ws + OFF_WR);
  _Float16*  Wm16 = (_Float16*)(ws + OFF_WM);

  // agg (6.4M floats) + deg (50k floats) are contiguous -> one zero pass
  int nz = N_NODES * NEMBED + N_NODES;
  k_zero<<<(nz + 255) / 256, 256, 0, stream>>>(agg, nz);

  k_cvt_w<<<(HDIM * NEMBED + 255) / 256, 256, 0, stream>>>(W_l, W_r, W_mlp, Wl16, Wr16, Wm16);

  // one wave per edge
  int scatter_blocks = (N_EDGES * 32) / 256;
  k_scatter<<<scatter_blocks, 256, 0, stream>>>(x, ei, agg, deg);

  int gemm_blocks = (N_NODES + MTILE - 1) / MTILE;
  k_sage_fused<<<gemm_blocks, 256, 0, stream>>>(x, agg, deg, Wl16, Wr16, Wm16,
                                                b_l, b_mlp, out);
}
